// GCN_83511344103768
// MI455X (gfx1250) — matrix operand order repaired and checked by reading, not compiled, not run on hardware
//
#include <hip/hip_runtime.h>
#include <hip/hip_bf16.h>

// ---------------------------------------------------------------------------
// GCN (3x GCNConv + global_add_pool) for MI455X / gfx1250, wave32.
// GEMMs: v_wmma_f32_16x16x32_bf16 (bf16 A/B, fp32 accumulate), each wave
// computes a 16x64 tile (4 WMMAs per K-step sharing one A fragment).
// W is pre-packed fragment-major so B fragments are two b128 loads.
// Sparse aggregation: fp32 global atomics (the memory-bound phase).
// ---------------------------------------------------------------------------

#define N_NODES  50000
#define N_EDGES  800000
#define IN_CH    128
#define HID      256
#define N_GRAPHS 512

typedef __attribute__((ext_vector_type(16))) __bf16 v16bf;
typedef __attribute__((ext_vector_type(8)))  float  v8f;
typedef __attribute__((ext_vector_type(8)))  unsigned short us8v;

// 32-byte fragment: two 16B halves in memory order == v16bf register order.
union frag32 {
    us8v  h[2];
    v16bf v;
};

__device__ __forceinline__ unsigned short f2bf_rne(float f) {
    union { float f; unsigned int u; } cv; cv.f = f;
    unsigned int u = cv.u;
    u += 0x7FFFu + ((u >> 16) & 1u);   // round-to-nearest-even
    return (unsigned short)(u >> 16);
}

// ---------------- degree / normalization ----------------
__global__ void k_degree(const int* __restrict__ dst, float* __restrict__ deg, int E) {
    int e = blockIdx.x * blockDim.x + threadIdx.x;
    if (e < E) atomicAdd(deg + dst[e], 1.0f);
}

__global__ void k_dinv(const float* __restrict__ deg, float* __restrict__ dinv, int N) {
    int i = blockIdx.x * blockDim.x + threadIdx.x;
    if (i < N) dinv[i] = rsqrtf(deg[i] + 1.0f);   // +1 = self loop
}

// ---------------- fp32 -> bf16 activation conversion ----------------
__global__ void k_cvt_bf16(const float* __restrict__ in, unsigned short* __restrict__ out, int n) {
    int i = blockIdx.x * blockDim.x + threadIdx.x;
    if (i < n) out[i] = f2bf_rne(in[i]);
}

// ---------------- pack W[K,256] fp32 -> fragment-major bf16 ----------------
// Wfrag[kt][half][col][j] (j=0..15 contiguous), kt = K/32 tiles.
// Element (kt,h,c,j) = W[(kt*32 + h*16 + j)*256 + c].
// A lane's whole B fragment = 32 contiguous bytes -> two global_load_b128.
__global__ void k_pack_w(const float* __restrict__ W, unsigned short* __restrict__ Wf, int K) {
    int i = blockIdx.x * blockDim.x + threadIdx.x;   // over K*256 elements
    if (i >= K * HID) return;
    int j  = i & 15;
    int c  = (i >> 4) & (HID - 1);
    int h  = (i >> 12) & 1;
    int kt = i >> 13;
    Wf[i] = f2bf_rne(W[(size_t)(kt * 32 + h * 16 + j) * HID + c]);
}

// ---------------- WMMA GEMM: H[M,256] = A[M,K] @ W[K,256] ----------------
// One wave computes a 16x64 tile: 4 WMMAs per 32-wide K-step, sharing one
// A fragment. A/B use contiguous k = half*16 + s (same order in both); f32 C 16x16 per CDNA5 ISA 7.12.2.
__global__ void k_gemm_wmma(const unsigned short* __restrict__ A,
                            const unsigned short* __restrict__ Wf,
                            float* __restrict__ H, int M, int K) {
    const int lane  = threadIdx.x & 31;
    const int wave  = threadIdx.x >> 5;
    const int half  = lane >> 4;      // 0: lanes 0-15, 1: lanes 16-31
    const int r     = lane & 15;
    const int mtile = blockIdx.x * 8 + wave;
    const int mbase = mtile * 16;
    if (mbase >= M) return;           // wave-uniform exit (EXEC stays all-1s)
    const int nbase = blockIdx.y * 64;
    const int row   = mbase + r;
    const bool rowok = row < M;

    v8f c0 = {}, c1 = {}, c2 = {}, c3 = {};
    for (int kk = 0; kk < K; kk += 32) {
        const int kt = kk >> 5;
        // A fragment: lane (half,r) = row r, K = {half*16..+15}
        frag32 fa;
        if (rowok) {
            const us8v* ap = (const us8v*)(A + (size_t)row * K + kk);
            fa.h[0] = ap[2 * half];      // kk + half*16       .. +7
            fa.h[1] = ap[2 * half + 1];  // kk + half*16 + 8   .. +7
        } else {
            fa.h[0] = (us8v)0; fa.h[1] = (us8v)0;
        }
        // 4 B fragments from fragment-major Wfrag: 2x b128 each
        #pragma unroll
        for (int t = 0; t < 4; ++t) {
            const int col = nbase + t * 16 + r;
            const us8v* bp =
                (const us8v*)(Wf + ((size_t)(kt * 2 + half) * HID + col) * 16);
            frag32 fb;
            fb.h[0] = bp[0];
            fb.h[1] = bp[1];
            v8f* cc = (t == 0) ? &c0 : (t == 1) ? &c1 : (t == 2) ? &c2 : &c3;
            *cc = __builtin_amdgcn_wmma_f32_16x16x32_bf16(
                      false, fa.v, false, fb.v, (short)0, *cc, false, false);
        }
    }
    // C tiles: VGPR v -> row mbase + half*8 + v, col nbase + t*16 + r
    #pragma unroll
    for (int v = 0; v < 8; ++v) {
        int orow = mbase + half * 8 + v;
        if (orow < M) {
            float* hp = H + (size_t)orow * HID + nbase + r;
            hp[0]  = c0[v];
            hp[16] = c1[v];
            hp[32] = c2[v];
            hp[48] = c3[v];
        }
    }
}

// ---------------- edge-parallel scatter-add (incl. self loops) ----------------
// work item = (edge-or-node, float4 feature chunk); 64 chunks of 4 per row.
__global__ void k_aggregate(const int* __restrict__ src, const int* __restrict__ dst,
                            const float* __restrict__ dinv,
                            const float* __restrict__ h, float* __restrict__ agg,
                            int E, int N) {
    long long idx = (long long)blockIdx.x * blockDim.x + threadIdx.x;
    long long total = (long long)(E + N) * 64;
    if (idx >= total) return;
    int e = (int)(idx >> 6);
    int c = (int)(idx & 63);
    int s, d;
    if (e < E) { s = src[e]; d = dst[e]; }
    else       { s = e - E;  d = s; }           // self loop
    float nrm = dinv[s] * dinv[d];
    const float4 v = *(const float4*)(h + (size_t)s * HID + c * 4);
    float* o = agg + (size_t)d * HID + c * 4;
    atomicAdd(o + 0, nrm * v.x);
    atomicAdd(o + 1, nrm * v.y);
    atomicAdd(o + 2, nrm * v.z);
    atomicAdd(o + 3, nrm * v.w);
}

// ---------------- bias + ReLU, write bf16 (next GEMM) and/or fp32 ----------------
__global__ void k_bias_relu(const float* __restrict__ agg, const float* __restrict__ bias,
                            unsigned short* __restrict__ out_bf, float* __restrict__ out_f,
                            int n) {
    int i = blockIdx.x * blockDim.x + threadIdx.x;
    if (i >= n) return;
    float v = agg[i] + bias[i & (HID - 1)];
    v = v > 0.0f ? v : 0.0f;
    if (out_bf) out_bf[i] = f2bf_rne(v);
    if (out_f)  out_f[i]  = v;
}

// ---------------- global_add_pool ----------------
__global__ void k_pool(const float* __restrict__ h, const int* __restrict__ batch,
                       float* __restrict__ out, int n) {
    int i = blockIdx.x * blockDim.x + threadIdx.x;
    if (i >= n) return;
    int node = i >> 8, f = i & (HID - 1);
    atomicAdd(out + (size_t)batch[node] * HID + f, h[i]);
}

// ---------------------------------------------------------------------------
extern "C" void kernel_launch(void* const* d_in, const int* in_sizes, int n_in,
                              void* d_out, int out_size, void* d_ws, size_t ws_size,
                              hipStream_t stream) {
    const float* x    = (const float*)d_in[0];
    const float* W0   = (const float*)d_in[1];
    const float* b0   = (const float*)d_in[2];
    const float* W1   = (const float*)d_in[3];
    const float* b1   = (const float*)d_in[4];
    const float* W2   = (const float*)d_in[5];
    const float* b2   = (const float*)d_in[6];
    const int*   eidx = (const int*)d_in[7];
    const int*   batch= (const int*)d_in[8];
    const int* src = eidx;
    const int* dst = eidx + N_EDGES;
    float* out = (float*)d_out;

    // workspace carve-up (256B aligned)
    char* ws = (char*)d_ws;
    size_t off = 0;
    auto carve = [&](size_t bytes) -> char* {
        char* p = ws + off;
        off = (off + bytes + 255) & ~(size_t)255;
        return p;
    };
    float*          deg  = (float*)carve((size_t)N_NODES * 4);
    float*          dinv = (float*)carve((size_t)N_NODES * 4);
    unsigned short* wbf  = (unsigned short*)carve((size_t)HID * HID * 2);
    unsigned short* abf  = (unsigned short*)carve((size_t)N_NODES * HID * 2);
    float*          hbuf = (float*)carve((size_t)N_NODES * HID * 4);
    float*          agg  = (float*)carve((size_t)N_NODES * HID * 4);

    const int T = 256;
    const size_t nh  = (size_t)N_NODES * HID;
    const long long aggWork = (long long)(N_EDGES + N_NODES) * 64;
    const dim3 gemmGrid((3125 + 7) / 8, HID / 64);   // ceil(50000/16) M-tiles / 8 waves, 4 N-supertiles

    // normalization (layer-independent)
    hipMemsetAsync(deg, 0, (size_t)N_NODES * 4, stream);
    k_degree<<<(N_EDGES + T - 1) / T, T, 0, stream>>>(dst, deg, N_EDGES);
    k_dinv<<<(N_NODES + T - 1) / T, T, 0, stream>>>(deg, dinv, N_NODES);

    // ---- layer 0: x[N,128] @ W0[128,256] ----
    k_cvt_bf16<<<((N_NODES * IN_CH) + T - 1) / T, T, 0, stream>>>(x, abf, N_NODES * IN_CH);
    k_pack_w<<<((IN_CH * HID) + T - 1) / T, T, 0, stream>>>(W0, wbf, IN_CH);
    k_gemm_wmma<<<gemmGrid, T, 0, stream>>>(abf, wbf, hbuf, N_NODES, IN_CH);
    hipMemsetAsync(agg, 0, nh * 4, stream);
    k_aggregate<<<(unsigned)((aggWork + T - 1) / T), T, 0, stream>>>(src, dst, dinv, hbuf, agg, N_EDGES, N_NODES);
    k_bias_relu<<<(unsigned)((nh + T - 1) / T), T, 0, stream>>>(agg, b0, abf, nullptr, (int)nh);

    // ---- layer 1: h[N,256] @ W1[256,256] ----
    k_pack_w<<<((HID * HID) + T - 1) / T, T, 0, stream>>>(W1, wbf, HID);
    k_gemm_wmma<<<gemmGrid, T, 0, stream>>>(abf, wbf, hbuf, N_NODES, HID);
    hipMemsetAsync(agg, 0, nh * 4, stream);
    k_aggregate<<<(unsigned)((aggWork + T - 1) / T), T, 0, stream>>>(src, dst, dinv, hbuf, agg, N_EDGES, N_NODES);
    k_bias_relu<<<(unsigned)((nh + T - 1) / T), T, 0, stream>>>(agg, b1, abf, nullptr, (int)nh);

    // ---- layer 2: h[N,256] @ W2[256,256] ----
    k_pack_w<<<((HID * HID) + T - 1) / T, T, 0, stream>>>(W2, wbf, HID);
    k_gemm_wmma<<<gemmGrid, T, 0, stream>>>(abf, wbf, hbuf, N_NODES, HID);
    hipMemsetAsync(agg, 0, nh * 4, stream);
    k_aggregate<<<(unsigned)((aggWork + T - 1) / T), T, 0, stream>>>(src, dst, dinv, hbuf, agg, N_EDGES, N_NODES);
    k_bias_relu<<<(unsigned)((nh + T - 1) / T), T, 0, stream>>>(agg, b2, nullptr, hbuf, (int)nh);

    // ---- global_add_pool ----
    hipMemsetAsync(out, 0, (size_t)N_GRAPHS * HID * 4, stream);
    k_pool<<<(unsigned)((nh + T - 1) / T), T, 0, stream>>>(hbuf, batch, out, (int)nh);
}
